// MultiHeadAttention_71073118814327
// MI455X (gfx1250) — matrix operand order repaired and checked by reading, compiled, not run on hardware
//
#include <hip/hip_runtime.h>

// MI455X / gfx1250 fused causal multi-head attention.
// f16 WMMA (v_wmma_f32_16x16x32_f16) everywhere, f32 accumulation.
// Round 3: __launch_bounds__ to kill scratch spills; lane-half exchanges via
// v_permlanex16 (VALU) instead of ds_bpermute (+s_wait_dscnt); packed-f16
// P fragment built from own lanes (no permlane exchange).

typedef __attribute__((ext_vector_type(16))) _Float16 v16h;
typedef __attribute__((ext_vector_type(8)))  _Float16 v8h;
typedef __attribute__((ext_vector_type(8)))  float    v8f;
typedef __attribute__((ext_vector_type(4)))  float    v4f;
typedef __attribute__((ext_vector_type(8)))  unsigned int v8u;

#define B_  4
#define S_  2048
#define D_  1024
#define H_  16
#define DK_ 64

__device__ __forceinline__ v8f wmma_f16(v16h a, v16h b, v8f c) {
    return __builtin_amdgcn_wmma_f32_16x16x32_f16(false, a, false, b, (short)0, c,
                                                  false, false);
}

// Swap lane halves 0-15 <-> 16-31 (identity selects): pure VALU, no DS pipe.
__device__ __forceinline__ unsigned permx16_u(unsigned x) {
    return (unsigned)__builtin_amdgcn_permlanex16(
        (int)x, (int)x, 0x76543210, 0xfedcba98, false, false);
}
__device__ __forceinline__ float permx16_f(float x) {
    return __builtin_bit_cast(float, permx16_u(__builtin_bit_cast(unsigned, x)));
}

// A-operand fragment (16x32, lane = row, K split 0..7 / 16..23 per lane group).
// Caller passes p = rowbase + k0 + 8*(lane>>4).
__device__ __forceinline__ v16h load_fragA_f16(const _Float16* p) {
    v8h lo = *(const v8h*)(p);
    v8h hi = *(const v8h*)(p + 16);
    v16h r;
#pragma unroll
    for (int i = 0; i < 8; ++i) { r[i] = lo[i]; r[i + 8] = hi[i]; }
    return r;
}

// Same A fragment but source is f32; convert inline (activations stay fp32 in HBM).
__device__ __forceinline__ v16h load_fragA_cvt_f32(const float* p) {
    v4f a0 = *(const v4f*)(p);
    v4f a1 = *(const v4f*)(p + 4);
    v4f b0 = *(const v4f*)(p + 16);
    v4f b1 = *(const v4f*)(p + 20);
    v16h r;
#pragma unroll
    for (int i = 0; i < 4; ++i) {
        r[i]      = (_Float16)a0[i];
        r[i + 4]  = (_Float16)a1[i];
        r[i + 8]  = (_Float16)b0[i];
        r[i + 12] = (_Float16)b1[i];
    }
    return r;
}

// B-operand fragment (32x16, lane = column, K split 0..7 / 16..23 per lane group).
// Caller passes p = colbase + k0 + 8*(lane>>4).
__device__ __forceinline__ v16h load_fragB_f16(const _Float16* p) {
    v8h lo = *(const v8h*)(p);
    v8h hi = *(const v8h*)(p + 16);
    v16h r;
#pragma unroll
    for (int i = 0; i < 8; ++i) { r[i] = lo[i]; r[i + 8] = hi[i]; }
    return r;
}

__device__ __forceinline__ v8f zero_v8f() {
    v8f z;
#pragma unroll
    for (int i = 0; i < 8; ++i) z[i] = 0.0f;
    return z;
}

// ---------------------------------------------------------------------------
// W[K][N] f32  ->  WT[N][K] f16   (LDS-tiled transpose, both sides coalesced)
// ---------------------------------------------------------------------------
__global__ __launch_bounds__(256) void transpose_cvt_kernel(
    const float* __restrict__ W, _Float16* __restrict__ WT) {
    __shared__ float tile[32][33];
    const int tx = threadIdx.x, ty = threadIdx.y;
    const int n0 = blockIdx.x * 32;
    const int k0 = blockIdx.y * 32;
#pragma unroll
    for (int i = 0; i < 4; ++i)
        tile[ty + i * 8][tx] = W[(size_t)(k0 + ty + i * 8) * D_ + (n0 + tx)];
    __syncthreads();
#pragma unroll
    for (int i = 0; i < 4; ++i)
        WT[(size_t)(n0 + ty + i * 8) * D_ + (k0 + tx)] =
            (_Float16)tile[tx][ty + i * 8];
}

// ---------------------------------------------------------------------------
// GEMM: C[M=8192][N=1024] = A[M][1024] x WT^T + bias
//   MODE 0: store f16 head-split  [B,H,S,DK]          (Q, K projections)
//   MODE 1: store f16 transposed  [B,H,DK,S]          (V projection)
//   MODE 2: store f32 row-major   [M][N]              (output projection)
// Wave tile: 16(M) x 64(N), 4 accumulators, K-step 32 -> 4 WMMA per step.
// Software pipelined: next step's A + B fragments load during current WMMAs.
// ---------------------------------------------------------------------------
template <int MODE, bool A_F32>
__global__ __launch_bounds__(128, 1) void gemm_kernel(
    const void* __restrict__ Av, const _Float16* __restrict__ WT,
    const float* __restrict__ bias, void* __restrict__ outv) {
    const int wave = threadIdx.x >> 5;
    const int lane = threadIdx.x & 31;
    const int lrow = lane & 15;
    const int lsel = lane >> 4;
    const int m0 = blockIdx.x * 64 + wave * 16;
    const int n0 = blockIdx.y * 64;

    v8f acc[4];
#pragma unroll
    for (int t = 0; t < 4; ++t) acc[t] = zero_v8f();

    const _Float16* wrow[4];
#pragma unroll
    for (int t = 0; t < 4; ++t)
        wrow[t] = WT + (size_t)(n0 + 16 * t + lrow) * D_ + 8 * lsel;

    if constexpr (A_F32) {
        const float* arow = (const float*)Av + (size_t)(m0 + lrow) * D_ + 8 * lsel;
        v16h a = load_fragA_cvt_f32(arow);
        v16h b[4];
#pragma unroll
        for (int t = 0; t < 4; ++t) b[t] = load_fragB_f16(wrow[t]);
        for (int k0 = 0; k0 < D_; k0 += 32) {
            v16h an = a;
            v16h bn[4] = {b[0], b[1], b[2], b[3]};
            const int kn = k0 + 32;
            if (kn < D_) {
                an = load_fragA_cvt_f32(arow + kn);
#pragma unroll
                for (int t = 0; t < 4; ++t) bn[t] = load_fragB_f16(wrow[t] + kn);
            }
#pragma unroll
            for (int t = 0; t < 4; ++t) acc[t] = wmma_f16(a, b[t], acc[t]);
            a = an;
#pragma unroll
            for (int t = 0; t < 4; ++t) b[t] = bn[t];
        }
    } else {
        const _Float16* arow =
            (const _Float16*)Av + (size_t)(m0 + lrow) * D_ + 8 * lsel;
        v16h a = load_fragA_f16(arow);
        v16h b[4];
#pragma unroll
        for (int t = 0; t < 4; ++t) b[t] = load_fragB_f16(wrow[t]);
        for (int k0 = 0; k0 < D_; k0 += 32) {
            v16h an = a;
            v16h bn[4] = {b[0], b[1], b[2], b[3]};
            const int kn = k0 + 32;
            if (kn < D_) {
                an = load_fragA_f16(arow + kn);
#pragma unroll
                for (int t = 0; t < 4; ++t) bn[t] = load_fragB_f16(wrow[t] + kn);
            }
#pragma unroll
            for (int t = 0; t < 4; ++t) acc[t] = wmma_f16(a, b[t], acc[t]);
            a = an;
#pragma unroll
            for (int t = 0; t < 4; ++t) b[t] = bn[t];
        }
    }

    // C/D layout: lane col n = lrow (+16t), rows m = r + 8*lsel.
    if constexpr (MODE == 0) {
        _Float16* dst = (_Float16*)outv;
#pragma unroll
        for (int t = 0; t < 4; ++t) {
            const int n = n0 + 16 * t + lrow;
            const int h = n >> 6, dk = n & 63;
            const float bv = bias[n];
#pragma unroll
            for (int r = 0; r < 8; ++r) {
                const int m = m0 + r + 8 * lsel;
                const int bidx = m >> 11, s = m & (S_ - 1);
                dst[((size_t)(bidx * H_ + h) * S_ + s) * DK_ + dk] =
                    (_Float16)(acc[t][r] + bv);
            }
        }
    } else if constexpr (MODE == 1) {
        _Float16* dst = (_Float16*)outv;
        const int mf = m0 + 8 * lsel;
        const int bidx = mf >> 11, s = mf & (S_ - 1);
#pragma unroll
        for (int t = 0; t < 4; ++t) {
            const int n = n0 + 16 * t + lrow;
            const int h = n >> 6, dk = n & 63;
            const float bv = bias[n];
            v8h pk;
#pragma unroll
            for (int r = 0; r < 8; ++r) pk[r] = (_Float16)(acc[t][r] + bv);
            *(v8h*)(dst + ((size_t)(bidx * H_ + h) * DK_ + dk) * S_ + s) = pk;
        }
    } else {
        float* dst = (float*)outv;
#pragma unroll
        for (int t = 0; t < 4; ++t) {
            const int n = n0 + 16 * t + lrow;
            const float bv = bias[n];
#pragma unroll
            for (int r = 0; r < 8; ++r) {
                const int m = m0 + r + 8 * lsel;
                dst[(size_t)m * D_ + n] = acc[t][r] + bv;
            }
        }
    }
}

// ---------------------------------------------------------------------------
// Flash attention (causal), transposed orientation:
//   S^T[key][q] = K(16k x 32dk) x Q^T(32dk x 16q)   -> lane column == query
//   O^T[dk][q]  = V^T(16dk x 32k) x P(32k x 16q)
// Per wave: 16 queries x full DK=64. 32-key blocks, online softmax in f32.
// K fragments software-pipelined; V loads hoisted ahead of the score WMMAs.
// All cross-lane traffic is v_permlanex16 (no LDS, no ds-waits).
// ---------------------------------------------------------------------------
__global__ __launch_bounds__(128, 1) void attn_kernel(
    const _Float16* __restrict__ Qh, const _Float16* __restrict__ Kh,
    const _Float16* __restrict__ VT, _Float16* __restrict__ Oc) {
    const int wave = threadIdx.x >> 5;
    const int lane = threadIdx.x & 31;
    const int lrow = lane & 15;
    const int lsel = lane >> 4;
    const int koff = 8 * lsel;

    const int bh = blockIdx.x >> 5;         // S_/64 = 32 query blocks per (b,h)
    const int qblk = blockIdx.x & 31;
    const int q0 = qblk * 64 + wave * 16;
    const int bidx = bh >> 4, h = bh & 15;

    const _Float16* Qb = Qh + (size_t)bh * S_ * DK_;
    const _Float16* Kb = Kh + (size_t)bh * S_ * DK_;
    const _Float16* Vb = VT + (size_t)bh * DK_ * S_;

    // Q^T as B fragments (column = this lane's query), dk halves 0..31 / 32..63
    v16h qb[2];
    qb[0] = load_fragB_f16(Qb + (size_t)(q0 + lrow) * DK_ + 0  + 8 * lsel);
    qb[1] = load_fragB_f16(Qb + (size_t)(q0 + lrow) * DK_ + 32 + 8 * lsel);

    v8f o[4];
#pragma unroll
    for (int t = 0; t < 4; ++t) o[t] = zero_v8f();
    float mstate = -3.0e38f, lsum = 0.0f;
    const int q = q0 + lrow;                // this lane's query index
    const int kb_end = q0 + 15;             // causal trip limit (inclusive)

    // K fragments: [0]=rows kb..kb+15 dk0..31, [1]=same dk32..63,
    //              [2]=rows kb+16..kb+31 dk0..31, [3]=same dk32..63
    v16h ka[4];
    auto load_ka = [&](int kb) {
        const _Float16* r0 = Kb + (size_t)(kb + lrow) * DK_ + koff;
        const _Float16* r1 = Kb + (size_t)(kb + 16 + lrow) * DK_ + koff;
        ka[0] = load_fragA_f16(r0);
        ka[1] = load_fragA_f16(r0 + 32);
        ka[2] = load_fragA_f16(r1);
        ka[3] = load_fragA_f16(r1 + 32);
    };
    load_ka(0);

    for (int kb = 0; kb <= kb_end; kb += 32) {
        // V fragments issue now; they complete under score-WMMA + softmax VALU.
        v16h va[4];
#pragma unroll
        for (int t = 0; t < 4; ++t)
            va[t] = load_fragA_f16(Vb + (size_t)(16 * t + lrow) * S_ + kb + koff);

        // ---- scores (transposed): rows = keys, cols = queries ----
        v8f s0 = zero_v8f(), s1 = zero_v8f();
        s0 = wmma_f16(ka[0], qb[0], s0);
        s0 = wmma_f16(ka[1], qb[1], s0);
        s1 = wmma_f16(ka[2], qb[0], s1);
        s1 = wmma_f16(ka[3], qb[1], s1);

        // Prefetch next block's K fragments under the softmax section.
        if (kb + 32 <= kb_end) load_ka(kb + 32);

        // ---- scale + causal mask + column(query)-max ----
        float mloc = -3.0e38f;
#pragma unroll
        for (int r = 0; r < 8; ++r) {
            const int kg0 = kb + r + koff;       // key of s0[r]
            const int kg1 = kg0 + 16;            // key of s1[r]
            const float x0 = (kg0 <= q) ? s0[r] * 0.125f : -3.0e38f;
            const float x1 = (kg1 <= q) ? s1[r] * 0.125f : -3.0e38f;
            s0[r] = x0; s1[r] = x1;
            mloc = fmaxf(mloc, fmaxf(x0, x1));
        }
        mloc = fmaxf(mloc, permx16_f(mloc));     // lanes l, l+16 share query q
        const float mnew = fmaxf(mstate, mloc);
        const float corr = __expf(mstate - mnew);
        mstate = mnew;

        // ---- P: exp, pack to f16 dwords, no lane swap needed ----
        // Own registers hold keys {0..7,16..23} (grp0) / {8..15,24..31} (grp1);
        // B fragment needs the same keys {0..7,16..23} / {8..15,24..31}.
        float e0[8], e1[8];
        float psum = 0.0f;
#pragma unroll
        for (int r = 0; r < 8; ++r) {
            e0[r] = __expf(s0[r] - mnew);
            e1[r] = __expf(s1[r] - mnew);
            psum += e0[r] + e1[r];
        }
        psum += permx16_f(psum);
        lsum = lsum * corr + psum;

        v8u pbu;
#pragma unroll
        for (int j = 0; j < 4; ++j) {
            const unsigned d0 = __builtin_bit_cast(
                unsigned, __builtin_amdgcn_cvt_pkrtz(e0[2 * j], e0[2 * j + 1]));
            const unsigned d1 = __builtin_bit_cast(
                unsigned, __builtin_amdgcn_cvt_pkrtz(e1[2 * j], e1[2 * j + 1]));
            const unsigned p0 = permx16_u(d0);   // partner's e0 pair
            const unsigned p1 = permx16_u(d1);   // partner's e1 pair
            pbu[j]     = d0;                     // keys 8*lsel + 2j..2j+1
            pbu[4 + j] = d1;                     // keys 16 + 8*lsel + 2j..2j+1
        }
        const v16h pb = __builtin_bit_cast(v16h, pbu);

#pragma unroll
        for (int t = 0; t < 4; ++t)
#pragma unroll
            for (int i = 0; i < 8; ++i) o[t][i] *= corr;

        // ---- O^T += V^T x P ----
#pragma unroll
        for (int t = 0; t < 4; ++t) o[t] = wmma_f16(va[t], pb, o[t]);
    }

    // ---- normalize, store context [B,S,D] f16 (8 contiguous dk per lane) ----
    const float rl = 1.0f / lsum;
    _Float16* dst = Oc + ((size_t)bidx * S_ + q) * D_ + h * DK_;
#pragma unroll
    for (int t = 0; t < 4; ++t) {
        v8h pk;
#pragma unroll
        for (int r = 0; r < 8; ++r) pk[r] = (_Float16)(o[t][r] * rl);
        *(v8h*)(dst + 16 * t + 8 * lsel) = pk;
    }
}

// ---------------------------------------------------------------------------
extern "C" void kernel_launch(void* const* d_in, const int* in_sizes, int n_in,
                              void* d_out, int out_size, void* d_ws,
                              size_t ws_size, hipStream_t stream) {
    const float* q  = (const float*)d_in[1];
    const float* k  = (const float*)d_in[2];
    const float* v  = (const float*)d_in[3];
    const float* wq = (const float*)d_in[5];
    const float* bq = (const float*)d_in[6];
    const float* wk = (const float*)d_in[7];
    const float* bk = (const float*)d_in[8];
    const float* wv = (const float*)d_in[9];
    const float* bv = (const float*)d_in[10];
    const float* wo = (const float*)d_in[11];
    const float* bo = (const float*)d_in[12];

    char* ws = (char*)d_ws;
    const size_t MB = 1ull << 20;
    _Float16* wqT = (_Float16*)(ws + 0 * MB);   // 2 MB each
    _Float16* wkT = (_Float16*)(ws + 2 * MB);
    _Float16* wvT = (_Float16*)(ws + 4 * MB);
    _Float16* woT = (_Float16*)(ws + 6 * MB);
    _Float16* Qh  = (_Float16*)(ws + 8 * MB);   // [B,H,S,DK]  16 MB
    _Float16* Kh  = (_Float16*)(ws + 24 * MB);  // [B,H,S,DK]  16 MB
    _Float16* VTb = (_Float16*)(ws + 40 * MB);  // [B,H,DK,S]  16 MB
    _Float16* Oc  = (_Float16*)(ws + 56 * MB);  // [B,S,D]     16 MB  (end 72 MB)

    dim3 tb(32, 8), tg(32, 32);
    transpose_cvt_kernel<<<tg, tb, 0, stream>>>(wq, wqT);
    transpose_cvt_kernel<<<tg, tb, 0, stream>>>(wk, wkT);
    transpose_cvt_kernel<<<tg, tb, 0, stream>>>(wv, wvT);
    transpose_cvt_kernel<<<tg, tb, 0, stream>>>(wo, woT);

    dim3 gb(128);                      // 4 waves per workgroup
    dim3 gg(128, 16);                  // M/64 = 128, N/64 = 16
    gemm_kernel<0, true><<<gg, gb, 0, stream>>>(q, wqT, bq, Qh);
    gemm_kernel<0, true><<<gg, gb, 0, stream>>>(k, wkT, bk, Kh);
    gemm_kernel<1, true><<<gg, gb, 0, stream>>>(v, wvT, bv, VTb);

    attn_kernel<<<dim3(B_ * H_ * (S_ / 64)), gb, 0, stream>>>(Qh, Kh, VTb, Oc);

    gemm_kernel<2, false><<<gg, gb, 0, stream>>>(Oc, woT, bo, d_out);
}
